// Block_29738353558238
// MI455X (gfx1250) — matrix operand order repaired and checked by reading, not compiled, not run on hardware
//
#include <hip/hip_runtime.h>

// MI455X / gfx1250, wave32. All GEMMs via v_wmma_f32_16x16x32_bf16.
// Global->LDS staging uses GLOBAL_LOAD_ASYNC_TO_LDS_B128 (ASYNCcnt) with
// double-buffered LDS so the next tile's fetch overlaps the current WMMAs.
// fp32 inputs are converted to bf16 (weights transposed to [N,K] so staging
// is k-contiguous); accumulation, softmax, layernorm, residuals stay fp32.

typedef __bf16 bf16;
typedef __attribute__((ext_vector_type(16))) __bf16 v16bf;
typedef __attribute__((ext_vector_type(8)))  float  v8f;
typedef unsigned int v4u __attribute__((ext_vector_type(4)));

union Frag32 { v4u u[2]; v16bf v; };

struct GemmP {
  const bf16* A; long lda, aB, aH;      // A base, leading dim, per-b / per-h offsets (elements)
  const bf16* B; long ldb, bB, bH;      // B base; if transB: element(k,n) at B[n*ldb + k]
  float* Cf; bf16* Cb; long ldc, cB, cH;
  const float* bias;                    // indexed by global column, may be null
  const float* res;                     // fp32 residual indexed row*ldc+col, may be null
  int M, N, K;
  int transB, gelu, causalSkip, causalK;
};

#define LDS_TILE (128 * 48)   // 12 KB per buffer (rows of 32 bf16 padded to 48)

// 128x128 tile per 256-thread block (8 waves as 4(M) x 2(N), each wave 32x64 = 2x4 frags).
__global__ __launch_bounds__(256) void gemm_bf16_wmma(GemmP p) {
  __shared__ bf16 As[2 * LDS_TILE];
  __shared__ bf16 Bs[2 * LDS_TILE];   // stored [n][k] so B fragments are K-contiguous

  const int tid  = threadIdx.x;
  const int lane = tid & 31;
  const int wid  = tid >> 5;
  const int wm   = wid & 3;
  const int wn   = wid >> 2;
  const long nBase = (long)blockIdx.x * 128;
  const long mBase = (long)blockIdx.y * 128;
  if (p.causalSkip && nBase > mBase + 127) return;   // fully-masked score block

  const long zb = blockIdx.z >> 2;   // batch (H = 4 hard-coded)
  const long zh = blockIdx.z & 3;    // head
  const bf16* Ag = p.A + zb * p.aB + zh * p.aH;
  const bf16* Bg = p.B + zb * p.bB + zh * p.bH;
  const long cOff = zb * p.cB + zh * p.cH;

  v8f acc[2][4];
  #pragma unroll
  for (int mi = 0; mi < 2; ++mi)
    #pragma unroll
    for (int ni = 0; ni < 4; ++ni)
      #pragma unroll
      for (int e = 0; e < 8; ++e) acc[mi][ni][e] = 0.0f;

  int kEnd = p.K;
  if (p.causalK) { long ke = mBase + 128; if (ke < kEnd) kEnd = (int)ke; }

  const int r  = lane & 15;   // A row / B col / C col within 16
  const int hf = lane >> 4;   // half-wave selector

  // Stage one 128x32 A tile + 32x128 B tile into LDS buffer `buf` for k-offset k0.
  auto stage = [&](int buf, int k0) {
    bf16* Asb = As + buf * LDS_TILE;
    bf16* Bsb = Bs + buf * LDS_TILE;
    { // A: 2 threads per row, 2x16B async copies each; IOFFSET applies to both sides
      int row = tid >> 1; int seg = (tid & 1) << 4;
      const bf16* g = Ag + (mBase + row) * p.lda + k0 + seg;
      unsigned la = (unsigned)(size_t)&Asb[row * 48 + seg];
      asm volatile(
          "global_load_async_to_lds_b128 %0, %1, off\n\t"
          "global_load_async_to_lds_b128 %0, %1, off offset:16"
          :: "v"(la), "v"(g) : "memory");
    }
    if (p.transB) { // element (k,n) at B[n*ldb+k]: k-contiguous, direct async copy
      int col = tid >> 1; int seg = (tid & 1) << 4;
      const bf16* g = Bg + (nBase + col) * p.ldb + k0 + seg;
      unsigned lb = (unsigned)(size_t)&Bsb[col * 48 + seg];
      asm volatile(
          "global_load_async_to_lds_b128 %0, %1, off\n\t"
          "global_load_async_to_lds_b128 %0, %1, off offset:16"
          :: "v"(lb), "v"(g) : "memory");
    } else {        // row-major [K,N]: coalesced load along n, transpose into Bs[n][k]
      int kk = tid >> 3; int seg = (tid & 7) << 4;
      const bf16* g = Bg + (long)(k0 + kk) * p.ldb + nBase + seg;
      v4u v0 = *(const v4u*)g;
      v4u v1 = *(const v4u*)(g + 8);
      const bf16* e0 = (const bf16*)&v0;
      const bf16* e1 = (const bf16*)&v1;
      #pragma unroll
      for (int i = 0; i < 8; ++i) Bsb[(seg + i) * 48 + kk]     = e0[i];
      #pragma unroll
      for (int i = 0; i < 8; ++i) Bsb[(seg + 8 + i) * 48 + kk] = e1[i];
    }
  };

  stage(0, 0);
  int cur = 0;
  for (int k0 = 0; k0 < kEnd; k0 += 32) {
    const bool hasNext = (k0 + 32 < kEnd);
    if (hasNext) stage(cur ^ 1, k0 + 32);    // prefetch into the other buffer

    // Wait until this wave's async copies for `cur` have landed (async loads
    // complete in order; the prefetch just issued is allowed to stay in flight).
    if (p.transB) {
      if (hasNext) asm volatile("s_wait_asynccnt 0x4" ::: "memory");
      else         asm volatile("s_wait_asynccnt 0x0" ::: "memory");
    } else {
      if (hasNext) asm volatile("s_wait_asynccnt 0x2" ::: "memory");
      else         asm volatile("s_wait_asynccnt 0x0" ::: "memory");
    }
    __syncthreads();   // cross-wave visibility of buffer `cur`

    const bf16* Asb = As + cur * LDS_TILE;
    const bf16* Bsb = Bs + cur * LDS_TILE;

    // Fragments per documented VGPR layouts:
    // A 16x32: lane holds row r, K = hf*8+{0..7} then 16+hf*8+{0..7}
    // B 32x16: lane holds col r, K = hf*8+{0..7} then 16+hf*8+{0..7}
    v16bf af[2], bfr[4];
    #pragma unroll
    for (int mi = 0; mi < 2; ++mi) {
      Frag32 f;
      int rowE = (wm * 32 + mi * 16 + r) * 48;
      f.u[0] = *(const v4u*)&Asb[rowE + hf * 8];
      f.u[1] = *(const v4u*)&Asb[rowE + 16 + hf * 8];
      af[mi] = f.v;
    }
    #pragma unroll
    for (int ni = 0; ni < 4; ++ni) {
      Frag32 f;
      int colE = (wn * 64 + ni * 16 + r) * 48;
      f.u[0] = *(const v4u*)&Bsb[colE + hf * 8];
      f.u[1] = *(const v4u*)&Bsb[colE + 16 + hf * 8];
      bfr[ni] = f.v;
    }

    #pragma unroll
    for (int mi = 0; mi < 2; ++mi)
      #pragma unroll
      for (int ni = 0; ni < 4; ++ni)
        acc[mi][ni] = __builtin_amdgcn_wmma_f32_16x16x32_bf16(
            false, af[mi], false, bfr[ni], (short)0, acc[mi][ni], false, false);

    __syncthreads();   // everyone done reading `cur` before it is restaged
    cur ^= 1;
  }

  // Epilogue. C layout: VGPR i -> M = hf*8+i, N = lane&15.
  #pragma unroll
  for (int mi = 0; mi < 2; ++mi)
    for (int ni = 0; ni < 4; ++ni) {
      long col = nBase + wn * 64 + ni * 16 + r;
      float bv = p.bias ? p.bias[col] : 0.0f;
      #pragma unroll
      for (int i = 0; i < 8; ++i) {
        long row = mBase + wm * 32 + mi * 16 + hf * 8 + i;
        float v = acc[mi][ni][i] + bv;
        if (p.res) v += p.res[row * p.ldc + col];
        if (p.gelu) {
          float t = v + 0.044715f * v * v * v;
          v = 0.5f * v * (1.0f + tanhf(0.7978845608028654f * t));
        }
        long idx = cOff + row * p.ldc + col;
        if (p.Cf) p.Cf[idx] = v;
        else      p.Cb[idx] = (bf16)v;
      }
    }
}

__global__ __launch_bounds__(256) void f32_to_bf16_k(const float* __restrict__ in,
                                                     bf16* __restrict__ out, long n) {
  long i = (long)blockIdx.x * 256 + threadIdx.x;
  long stride = (long)gridDim.x * 256;
  for (; i < n; i += stride) out[i] = (bf16)in[i];
}

// Convert fp32 [K,N] row-major -> bf16 [N,K] row-major (transposed weight).
// K is a power of two (kShift); writes coalesced, reads strided.
__global__ __launch_bounds__(256) void f32_to_bf16_tr_k(const float* __restrict__ in,
                                                        bf16* __restrict__ out,
                                                        int kShift, long N, long n) {
  long i = (long)blockIdx.x * 256 + threadIdx.x;
  long stride = (long)gridDim.x * 256;
  long kMask = (1l << kShift) - 1;
  for (; i < n; i += stride) {
    long n0 = i >> kShift;
    long k  = i & kMask;
    out[i] = (bf16)in[k * N + n0];
  }
}

// Causal softmax over one score row; scores are unscaled q.k, scale applied here.
__global__ __launch_bounds__(256) void softmax_causal_k(const float* __restrict__ sc,
                                                        bf16* __restrict__ pr, int S) {
  const int row = blockIdx.x;
  const long z  = blockIdx.y;
  const float scale = 0.044194173824159216f;  // 1/sqrt(512)
  const float* s = sc + (z * S + row) * (long)S;
  bf16* o = pr + (z * S + row) * (long)S;
  __shared__ float red[256];

  float mx = -3.0e38f;
  for (int c = threadIdx.x; c <= row; c += 256) mx = fmaxf(mx, s[c] * scale);
  red[threadIdx.x] = mx; __syncthreads();
  for (int off = 128; off; off >>= 1) {
    if ((int)threadIdx.x < off) red[threadIdx.x] = fmaxf(red[threadIdx.x], red[threadIdx.x + off]);
    __syncthreads();
  }
  mx = red[0]; __syncthreads();

  float sum = 0.0f;
  for (int c = threadIdx.x; c <= row; c += 256) sum += expf(s[c] * scale - mx);
  red[threadIdx.x] = sum; __syncthreads();
  for (int off = 128; off; off >>= 1) {
    if ((int)threadIdx.x < off) red[threadIdx.x] += red[threadIdx.x + off];
    __syncthreads();
  }
  float inv = 1.0f / red[0];

  for (int c = threadIdx.x; c < S; c += 256)
    o[c] = (c <= row) ? (bf16)(expf(s[c] * scale - mx) * inv) : (bf16)0.0f;
}

__global__ __launch_bounds__(256) void layernorm_k(const float* __restrict__ in,
                                                   const float* __restrict__ g,
                                                   const float* __restrict__ b,
                                                   float* __restrict__ of,
                                                   bf16* __restrict__ ob, int N) {
  long row = blockIdx.x;
  const float* xr = in + row * (long)N;
  __shared__ float red[256];

  float s = 0.0f;
  for (int c = threadIdx.x; c < N; c += 256) s += xr[c];
  red[threadIdx.x] = s; __syncthreads();
  for (int off = 128; off; off >>= 1) {
    if ((int)threadIdx.x < off) red[threadIdx.x] += red[threadIdx.x + off];
    __syncthreads();
  }
  float mean = red[0] / N; __syncthreads();

  float v = 0.0f;
  for (int c = threadIdx.x; c < N; c += 256) { float d = xr[c] - mean; v += d * d; }
  red[threadIdx.x] = v; __syncthreads();
  for (int off = 128; off; off >>= 1) {
    if ((int)threadIdx.x < off) red[threadIdx.x] += red[threadIdx.x + off];
    __syncthreads();
  }
  float inv = rsqrtf(red[0] / N + 1e-5f);

  for (int c = threadIdx.x; c < N; c += 256) {
    float y = g[c] * (xr[c] - mean) * inv + b[c];
    if (of) of[row * (long)N + c] = y;
    if (ob) ob[row * (long)N + c] = (bf16)y;
  }
}

// out[row*2560 + col] = x[row*2048 + col]  (grow_head concat of x into slots 0..3)
__global__ __launch_bounds__(256) void concat_x_k(const float* __restrict__ x,
                                                  float* __restrict__ out, long n) {
  long i = (long)blockIdx.x * 256 + threadIdx.x;
  long stride = (long)gridDim.x * 256;
  for (; i < n; i += stride) {
    long row = i >> 11;        // /2048
    long col = i & 2047;
    out[row * 2560 + col] = x[i];
  }
}

extern "C" void kernel_launch(void* const* d_in, const int* in_sizes, int n_in,
                              void* d_out, int out_size, void* d_ws, size_t ws_size,
                              hipStream_t stream) {
  const float* x     = (const float*)d_in[0];
  const float* w_qkv = (const float*)d_in[1];
  const float* b_qkv = (const float*)d_in[2];
  const float* w_ao  = (const float*)d_in[3];
  const float* b_ao  = (const float*)d_in[4];
  const float* ln1_g = (const float*)d_in[5];
  const float* ln1_b = (const float*)d_in[6];
  const float* w_fc  = (const float*)d_in[7];
  const float* b_fc  = (const float*)d_in[8];
  const float* w_pr  = (const float*)d_in[9];
  const float* b_pr  = (const float*)d_in[10];
  const float* ln2_g = (const float*)d_in[11];
  const float* ln2_b = (const float*)d_in[12];
  const float* w_out = (const float*)d_in[13];
  const float* b_out = (const float*)d_in[14];
  float* out = (float*)d_out;

  const long Bn = 4, S = 2048, NX = 2048, E = 512, Hh = 4, M0 = Bn * S;
  const long MiB = 1ll << 20;
  char* ws = (char*)d_ws;
  // Slab arena (lifetime reuse), 544 MiB total:
  bf16*  Wb   = (bf16*) (ws + 0 * MiB);     // current transposed bf16 weight (<=32MiB, reused)
  bf16*  ACT  = (bf16*) (ws + 32 * MiB);    // xb -> attn_out -> n_bf16 -> h_bf16 (32MiB)
  bf16*  QKVb = (bf16*) (ws + 64 * MiB);    // qkv bf16 (96MiB)
  float* SC   = (float*)(ws + 160 * MiB);   // scores fp32 (256MiB)
  bf16*  Gb   = (bf16*) (ws + 160 * MiB);   // gelu(fc) bf16 (128MiB, reuses SC)
  bf16*  PBp  = (bf16*) (ws + 416 * MiB);   // probs bf16 (128MiB)
  float* PRE  = (float*)(ws + 416 * MiB);   // pre-LN fp32 (64MiB, reuses probs)
  float* NF   = (float*)(ws + 480 * MiB);   // n fp32 residual (64MiB)

  auto conv = [&](const float* src, bf16* dst, long n) {
    f32_to_bf16_k<<<4096, 256, 0, stream>>>(src, dst, n);
  };
  auto convT = [&](const float* src, bf16* dst, int kShift, long N, long n) {
    f32_to_bf16_tr_k<<<4096, 256, 0, stream>>>(src, dst, kShift, N, n);
  };
  auto gemm = [&](const GemmP& p, int batch) {
    dim3 grid((unsigned)(p.N / 128), (unsigned)(p.M / 128), (unsigned)batch);
    gemm_bf16_wmma<<<grid, 256, 0, stream>>>(p);
  };

  // 1) x -> bf16
  conv(x, ACT, M0 * NX);
  // 2) qkv = x @ w_qkv + b_qkv  (bf16 out); weight transposed -> async transB path
  convT(w_qkv, Wb, 11, 3 * NX, NX * 3 * NX);
  { GemmP p{}; p.A = ACT; p.lda = NX; p.B = Wb; p.ldb = NX; p.transB = 1;
    p.Cb = QKVb; p.ldc = 3 * NX; p.bias = b_qkv;
    p.M = (int)M0; p.N = (int)(3 * NX); p.K = (int)NX; gemm(p, 1); }
  // 3) scores = q @ k^T per (b,h), skipping fully-masked blocks
  { GemmP p{}; p.A = QKVb; p.lda = 3 * NX; p.aB = S * 3 * NX; p.aH = E;
    p.B = QKVb + NX; p.ldb = 3 * NX; p.bB = S * 3 * NX; p.bH = E; p.transB = 1;
    p.Cf = SC; p.ldc = S; p.cB = Hh * S * S; p.cH = S * S;
    p.M = (int)S; p.N = (int)S; p.K = (int)E; p.causalSkip = 1;
    gemm(p, (int)(Bn * Hh)); }
  // 4) causal softmax -> bf16 probs
  softmax_causal_k<<<dim3((unsigned)S, (unsigned)(Bn * Hh)), 256, 0, stream>>>(SC, PBp, (int)S);
  // 5) a = probs @ v per (b,h), K clamped at the diagonal (v row-major: scatter path)
  { GemmP p{}; p.A = PBp; p.lda = S; p.aB = Hh * S * S; p.aH = S * S;
    p.B = QKVb + 2 * NX; p.ldb = 3 * NX; p.bB = S * 3 * NX; p.bH = E;
    p.Cb = ACT; p.ldc = NX; p.cB = S * NX; p.cH = E;
    p.M = (int)S; p.N = (int)E; p.K = (int)S; p.causalK = 1;
    gemm(p, (int)(Bn * Hh)); }
  // 6) pre_ln1 = a @ w_ao + b_ao + x  (fp32)
  convT(w_ao, Wb, 11, NX, NX * NX);
  { GemmP p{}; p.A = ACT; p.lda = NX; p.B = Wb; p.ldb = NX; p.transB = 1;
    p.Cf = PRE; p.ldc = NX; p.bias = b_ao; p.res = x;
    p.M = (int)M0; p.N = (int)NX; p.K = (int)NX; gemm(p, 1); }
  // 7) n = LN1(pre_ln1): fp32 copy for residual + bf16 copy for FC
  layernorm_k<<<(unsigned)M0, 256, 0, stream>>>(PRE, ln1_g, ln1_b, NF, ACT, (int)NX);
  // 8) g = gelu(n @ w_fc + b_fc)  (bf16)
  convT(w_fc, Wb, 11, 4 * NX, NX * 4 * NX);
  { GemmP p{}; p.A = ACT; p.lda = NX; p.B = Wb; p.ldb = NX; p.transB = 1;
    p.Cb = Gb; p.ldc = 4 * NX; p.bias = b_fc; p.gelu = 1;
    p.M = (int)M0; p.N = (int)(4 * NX); p.K = (int)NX; gemm(p, 1); }
  // 9) pre_ln2 = g @ w_pr + b_pr + n  (fp32); w_pr is [8192,2048] -> kShift 13
  convT(w_pr, Wb, 13, NX, 4 * NX * NX);
  { GemmP p{}; p.A = Gb; p.lda = 4 * NX; p.B = Wb; p.ldb = 4 * NX; p.transB = 1;
    p.Cf = PRE; p.ldc = NX; p.bias = b_pr; p.res = NF;
    p.M = (int)M0; p.N = (int)NX; p.K = (int)(4 * NX); gemm(p, 1); }
  // 10) h = LN2(pre_ln2) -> bf16 only
  layernorm_k<<<(unsigned)M0, 256, 0, stream>>>(PRE, ln2_g, ln2_b, nullptr, ACT, (int)NX);
  // 11) out tail = h @ w_out + b_out, written strided into d_out[:, 2048:2560]
  convT(w_out, Wb, 11, E, NX * E);
  { GemmP p{}; p.A = ACT; p.lda = NX; p.B = Wb; p.ldb = NX; p.transB = 1;
    p.Cf = out + NX; p.ldc = (Hh + 1) * E; p.bias = b_out;
    p.M = (int)M0; p.N = (int)E; p.K = (int)NX; gemm(p, 1); }
  // 12) concat x into d_out[:, 0:2048]
  concat_x_k<<<8192, 256, 0, stream>>>(x, out, M0 * NX);
}
